// LowRankSparseAttention_82240033784125
// MI455X (gfx1250) — matrix operand order repaired and checked by reading, not compiled, not run on hardware
//
#include <hip/hip_runtime.h>
#include <math.h>

typedef __attribute__((ext_vector_type(16))) _Float16 v16h;
typedef __attribute__((ext_vector_type(8)))  _Float16 v8h;
typedef __attribute__((ext_vector_type(8)))  float    v8f;

#define D_MODEL 1024
#define N_HEADS 16
#define RANK    128
#define TOPKN   64
#define D_HEAD  64
#define SEQLEN  2048
#define BATCH   2
#define SCALE_F 0.125f   // 64^-0.5

// ---------------- wave helpers (wave32) ----------------
__device__ __forceinline__ float wred_add_f(float v) {
  #pragma unroll
  for (int o = 16; o > 0; o >>= 1) v += __shfl_xor(v, o);
  return v;
}
__device__ __forceinline__ unsigned wred_max_u(unsigned v) {
  #pragma unroll
  for (int o = 16; o > 0; o >>= 1) { unsigned t = __shfl_xor(v, o); v = (t > v) ? t : v; }
  return v;
}
// exclusive scan over 32 lanes; also returns wave total
__device__ __forceinline__ int wexcl_scan(int v, int lane, int& total) {
  int incl = v;
  #pragma unroll
  for (int o = 1; o < 32; o <<= 1) { int y = __shfl_up(incl, o); if (lane >= o) incl += y; }
  total = __shfl(incl, 31);
  return incl - v;
}

// monotone u16 key for f16 bits: bigger float -> bigger key
__device__ __forceinline__ unsigned mono16(unsigned h) {
  return h ^ ((h & 0x8000u) ? 0xFFFFu : 0x8000u);
}
__device__ __forceinline__ float key_to_f(unsigned k) {
  unsigned h = (k & 0x8000u) ? (k ^ 0x8000u) : (k ^ 0xFFFFu);
  _Float16 hf = __builtin_bit_cast(_Float16, (unsigned short)h);
  return (float)hf;
}

// From a 256-bin per-row LDS histogram, find largest byte b with
// suffix-count(>=b) >= k.  Returns b, S(b)=count(>=b), H(b)=hist[b].
// Whole wave cooperates; hist owned exclusively by this wave (LDS in-order).
__device__ __forceinline__ void hist_select(const unsigned* hist, int k, int lane,
                                            int& bsel, int& Sb, int& Hb) {
  uint4 h0 = ((const uint4*)hist)[lane * 2];
  uint4 h1 = ((const uint4*)hist)[lane * 2 + 1];
  unsigned h[8] = { h0.x, h0.y, h0.z, h0.w, h1.x, h1.y, h1.z, h1.w };
  int s = 0;
  #pragma unroll
  for (int j = 0; j < 8; ++j) s += (int)h[j];
  int total;
  int excl = wexcl_scan(s, lane, total);
  int running = total - excl - s;       // suffix over lanes above this one
  int lb = -1, lS = 0, lH = 0;
  #pragma unroll
  for (int j = 7; j >= 0; --j) {
    running += (int)h[j];
    if (running >= k && lb < 0) { lb = lane * 8 + j; lS = running; lH = (int)h[j]; }
  }
  int b = lb;
  #pragma unroll
  for (int o = 16; o > 0; o >>= 1) { int t = __shfl_xor(b, o); b = (t > b) ? t : b; }
  bsel = b;
  Sb = __shfl(lS, b >> 3);
  Hb = __shfl(lH, b >> 3);
}

// ---------------- elementwise f32 -> f16 cast ----------------
__global__ void cast_f32_f16(const float* __restrict__ in, _Float16* __restrict__ out, int n) {
  int i = blockIdx.x * blockDim.x + threadIdx.x;
  if (i < n) out[i] = (_Float16)in[i];
}

// in: [D][R] f32 row-major  ->  out: [R][D] f16 (transposed cast)
__global__ void transpose_cast_f32_f16(const float* __restrict__ in,
                                       _Float16* __restrict__ out, int D, int R) {
  int o = blockIdx.x * blockDim.x + threadIdx.x;
  if (o < D * R) {
    int r = o / D, d = o - r * D;
    out[o] = (_Float16)in[d * R + r];
  }
}

// ---------------- generic WMMA f16 GEMM (B pre-transposed) ----------------
// One wave per 16x16 output tile.  D = A[M,K] * B[K,N] (+bias), f32 accumulate.
// A row-major (lda).  Bt[N][ldb]: row n holds B column n, K-contiguous.
__global__ __launch_bounds__(32) void gemm_wmma16(
    const _Float16* __restrict__ A, int lda,
    const _Float16* __restrict__ Bt, int ldb,
    int K,
    const float* __restrict__ bias,
    _Float16* __restrict__ outH, float* __restrict__ outF, int ldc) {
  const int lane = threadIdx.x;
  const int half = lane >> 4;      // K-half select
  const int nr   = lane & 15;      // A-row / B-col / D-col
  const int m0 = blockIdx.y * 16;
  const int n0 = blockIdx.x * 16;

  v8f c = {};
  const _Float16* arow = A  + (long)(m0 + nr) * lda;
  const _Float16* brow = Bt + (long)(n0 + nr) * ldb;
  for (int k0 = 0; k0 < K; k0 += 32) {
    v8h alo = *(const v8h*)(arow + k0 + half * 8);        // K = h*8 + e
    v8h ahi = *(const v8h*)(arow + k0 + 16 + half * 8);   // K = 16 + h*8 + e
    v16h a  = __builtin_shufflevector(alo, ahi, 0,1,2,3,4,5,6,7,8,9,10,11,12,13,14,15);
    v8h blo = *(const v8h*)(brow + k0 + half * 8);        // K = h*8 + e
    v8h bhi = *(const v8h*)(brow + k0 + 16 + half * 8);   // K = 16 + h*8 + e
    v16h bf = __builtin_shufflevector(blo, bhi, 0,1,2,3,4,5,6,7,8,9,10,11,12,13,14,15);
    c = __builtin_amdgcn_wmma_f32_16x16x32_f16(false, a, false, bf,
                                               (short)0, c, false, false);
  }
  const float bv = bias ? bias[n0 + nr] : 0.0f;
  if (outH) {
    #pragma unroll
    for (int r = 0; r < 8; ++r)
      outH[(long)(m0 + half * 8 + r) * ldc + n0 + nr] = (_Float16)(c[r] + bv);
  } else {
    #pragma unroll
    for (int r = 0; r < 8; ++r)
      outF[(long)(m0 + half * 8 + r) * ldc + n0 + nr] = c[r] + bv;
  }
}

// ---------------- fused scores + top-k softmax + sparse AV ----------------
// Grid: BATCH * N_HEADS * (SEQLEN/16) blocks of 256 threads (8 waves).
// Dynamic LDS: f16 scores[16][2048] = 65536 B.  Per-row buffer (4 KB) is reused
// after its keys are register-cached: [0,256)B selP, [256,384)B selI,
// [1024,2048)B 256-bin histogram.
__global__ __launch_bounds__(256) void attn_topk_kernel(
    const _Float16* __restrict__ yq,   // [B, L, D_MODEL] f16 (head h at col h*64)
    const _Float16* __restrict__ yk,
    const _Float16* __restrict__ yv,
    _Float16* __restrict__ attO) {     // [B, L, D_MODEL] f16
  extern __shared__ char smem[];
  _Float16* sc = (_Float16*)smem;      // [16][SEQLEN]

  const int tile = blockIdx.x;
  const int nqt  = SEQLEN / 16;
  const int qt   = tile % nqt;
  const int h    = (tile / nqt) % N_HEADS;
  const int b    = tile / (nqt * N_HEADS);
  const int m0   = qt * 16;

  const int wave = threadIdx.x >> 5;
  const int lane = threadIdx.x & 31;
  const int half = lane >> 4;
  const int nr   = lane & 15;

  // ---- Phase 1: score strip via WMMA (each wave: 16 key tiles) ----
  const _Float16* qrow = yq + ((long)(b * SEQLEN + m0 + nr) * D_MODEL + h * D_HEAD);
  v8h q0 = *(const v8h*)(qrow + half * 8);
  v8h q1 = *(const v8h*)(qrow + 16 + half * 8);
  v8h q2 = *(const v8h*)(qrow + 32 + half * 8);
  v8h q3 = *(const v8h*)(qrow + 48 + half * 8);
  v16h aq0 = __builtin_shufflevector(q0, q1, 0,1,2,3,4,5,6,7,8,9,10,11,12,13,14,15);
  v16h aq1 = __builtin_shufflevector(q2, q3, 0,1,2,3,4,5,6,7,8,9,10,11,12,13,14,15);
  for (int i = 0; i < 16; ++i) {
    const int kt  = wave + i * 8;            // 0..127
    const int key = kt * 16 + nr;
    const _Float16* krow = yk + ((long)(b * SEQLEN + key) * D_MODEL + h * D_HEAD);
    v16h b0 = __builtin_shufflevector(*(const v8h*)(krow + half * 8), *(const v8h*)(krow + 16 + half * 8), 0,1,2,3,4,5,6,7,8,9,10,11,12,13,14,15);        // K = h*8+e | 16+h*8+e, chunk 0
    v16h b1 = __builtin_shufflevector(*(const v8h*)(krow + 32 + half * 8), *(const v8h*)(krow + 48 + half * 8), 0,1,2,3,4,5,6,7,8,9,10,11,12,13,14,15);   // chunk 1
    v8f c = {};
    c = __builtin_amdgcn_wmma_f32_16x16x32_f16(false, aq0, false, b0, (short)0, c, false, false);
    c = __builtin_amdgcn_wmma_f32_16x16x32_f16(false, aq1, false, b1, (short)0, c, false, false);
    #pragma unroll
    for (int r = 0; r < 8; ++r)
      sc[(half * 8 + r) * SEQLEN + key] = (_Float16)(c[r] * SCALE_F);
  }
  __syncthreads();

  // ---- Phase 2: per-row histogram top-64, softmax, sparse AV ----
  const unsigned* vbase32 =
      (const unsigned*)(yv + (long)b * SEQLEN * D_MODEL + h * D_HEAD);
  for (int rr = 0; rr < 2; ++rr) {
    const int m = wave + rr * 8;             // row 0..15 owned by this wave

    // register-cache this lane's 64-score chunk (j = lane*64 .. +63), 2/dword
    const unsigned* rowu = (const unsigned*)sc + m * (SEQLEN / 2) + lane * 32;
    unsigned kreg[32];
    #pragma unroll
    for (int t = 0; t < 32; ++t) {
      unsigned x  = rowu[t];
      kreg[t] = mono16(x & 0xFFFFu) | (mono16(x >> 16) << 16);
    }

    // row max key
    unsigned mx = 0;
    #pragma unroll
    for (int t = 0; t < 32; ++t) {
      unsigned lo = kreg[t] & 0xFFFFu, hi = kreg[t] >> 16;
      unsigned mt = (lo > hi) ? lo : hi;
      if (mt > mx) mx = mt;
    }
    mx = wred_max_u(mx);

    // -- level 1: 256-bin histogram over key high byte --
    unsigned* hist = (unsigned*)(sc + (long)m * SEQLEN + 512);   // row offset 1024 B
    ((uint4*)hist)[lane * 2]     = make_uint4(0, 0, 0, 0);
    ((uint4*)hist)[lane * 2 + 1] = make_uint4(0, 0, 0, 0);
    #pragma unroll
    for (int t = 0; t < 32; ++t) {
      unsigned x = kreg[t];
      atomicAdd(&hist[(x >> 8) & 0xFFu], 1u);
      atomicAdd(&hist[x >> 24], 1u);
    }
    int hb, Shb, Hhb;
    hist_select(hist, TOPKN, lane, hb, Shb, Hhb);
    const int cgt_hi = Shb - Hhb;            // #(key high byte > hb), < 64
    const int k2 = TOPKN - cgt_hi;           // rank within hb bucket, >= 1

    // -- level 2: low-byte histogram among keys with high byte == hb --
    ((uint4*)hist)[lane * 2]     = make_uint4(0, 0, 0, 0);
    ((uint4*)hist)[lane * 2 + 1] = make_uint4(0, 0, 0, 0);
    #pragma unroll
    for (int t = 0; t < 32; ++t) {
      unsigned lo = kreg[t] & 0xFFFFu, hi = kreg[t] >> 16;
      if ((int)(lo >> 8) == hb) atomicAdd(&hist[lo & 0xFFu], 1u);
      if ((int)(hi >> 8) == hb) atomicAdd(&hist[hi & 0xFFu], 1u);
    }
    int lbyte, S2, H2;
    hist_select(hist, k2, lane, lbyte, S2, H2);
    const unsigned T  = ((unsigned)hb << 8) | (unsigned)lbyte;   // 64th largest key
    const int ngt      = cgt_hi + (S2 - H2); // #(key > T), < 64
    const int neq_need = TOPKN - ngt;        // ties kept, lowest index first

    // per-lane counts for compaction ranks
    int cgt = 0, ceq = 0;
    #pragma unroll
    for (int t = 0; t < 32; ++t) {
      unsigned lo = kreg[t] & 0xFFFFu, hi = kreg[t] >> 16;
      cgt += (lo > T) + (hi > T);
      ceq += (lo == T) + (hi == T);
    }
    int tot_gt, tot_eq;
    const int gtx = wexcl_scan(cgt, lane, tot_gt);
    const int eqx = wexcl_scan(ceq, lane, tot_eq);
    (void)tot_gt; (void)tot_eq;

    // compact (unnormalized e, idx) into LDS, reusing this row's score space
    const float vmaxf = key_to_f(mx);
    const float eT    = __expf(key_to_f(T) - vmaxf);
    float*          selP = (float*)(sc + (long)m * SEQLEN);      // 64 f32
    unsigned short* selI = (unsigned short*)(selP + TOPKN);      // 64 u16
    int lgt = 0, leq = 0;
    #pragma unroll
    for (int t = 0; t < 32; ++t) {
      #pragma unroll
      for (int s = 0; s < 2; ++s) {
        const unsigned kv = (s == 0) ? (kreg[t] & 0xFFFFu) : (kreg[t] >> 16);
        const int j = lane * 64 + 2 * t + s;
        if (kv > T) {
          const int pos = gtx + lgt++;
          selP[pos] = __expf(key_to_f(kv) - vmaxf);
          selI[pos] = (unsigned short)j;
        } else if (kv == T) {
          const int er = eqx + leq++;
          if (er < neq_need) {
            const int pos = ngt + er;
            selP[pos] = eT;
            selI[pos] = (unsigned short)j;
          }
        }
      }
    }

    // softmax denominator from the 64 compacted values
    const float sum = wred_add_f(selP[lane] + selP[lane + 32]);
    const float inv_sum = 1.0f / sum;

    // sparse AV: lane covers head dims {2*lane, 2*lane+1}; one b32 per key
    float a0 = 0.0f, a1 = 0.0f;
    for (int t = 0; t < TOPKN; ++t) {
      const float p = selP[t];
      const int   j = selI[t];
      const unsigned pv = vbase32[(j << 9) + lane];   // 512 dwords per key row
      const _Float16 v0 = __builtin_bit_cast(_Float16, (unsigned short)(pv & 0xFFFFu));
      const _Float16 v1 = __builtin_bit_cast(_Float16, (unsigned short)(pv >> 16));
      a0 += p * (float)v0;
      a1 += p * (float)v1;
    }
    const unsigned short o0 = __builtin_bit_cast(unsigned short, (_Float16)(a0 * inv_sum));
    const unsigned short o1 = __builtin_bit_cast(unsigned short, (_Float16)(a1 * inv_sum));
    unsigned* orow32 =
        (unsigned*)(attO + ((long)(b * SEQLEN + m0 + m) * D_MODEL + h * D_HEAD));
    orow32[lane] = (unsigned)o0 | ((unsigned)o1 << 16);
  }
}

// ---------------- host side ----------------
static inline void run_cast(const float* in, _Float16* out, int n, hipStream_t s) {
  cast_f32_f16<<<(n + 255) / 256, 256, 0, s>>>(in, out, n);
}

extern "C" void kernel_launch(void* const* d_in, const int* in_sizes, int n_in,
                              void* d_out, int out_size, void* d_ws, size_t ws_size,
                              hipStream_t stream) {
  (void)in_sizes; (void)n_in; (void)out_size;
  const float* q  = (const float*)d_in[0];
  const float* k  = (const float*)d_in[1];
  const float* v  = (const float*)d_in[2];
  const float* U[4] = { (const float*)d_in[3], (const float*)d_in[6],
                        (const float*)d_in[9], (const float*)d_in[12] };
  const float* V[4] = { (const float*)d_in[4], (const float*)d_in[7],
                        (const float*)d_in[10], (const float*)d_in[13] };
  const float* bia[4] = { (const float*)d_in[5], (const float*)d_in[8],
                          (const float*)d_in[11], (const float*)d_in[14] };
  float* out = (float*)d_out;

  const long ROWS = (long)BATCH * SEQLEN;          // 4096
  const long XEL  = ROWS * D_MODEL;                // 4M elements
  const long FEL  = (long)D_MODEL * RANK;          // 128K elements

  char* ws = (char*)d_ws;
  size_t off = 0;
  auto take = [&](size_t bytes) -> char* {
    char* p = ws + off;
    off += (bytes + 255) & ~(size_t)255;
    return p;
  };
  _Float16* x16[3];
  for (int i = 0; i < 3; ++i) x16[i] = (_Float16*)take(XEL * 2);
  _Float16 *UT16[4], *V16[4];
  for (int i = 0; i < 4; ++i) { UT16[i] = (_Float16*)take(FEL * 2); V16[i] = (_Float16*)take(FEL * 2); }
  _Float16* tmp16 = (_Float16*)take(ROWS * RANK * 2);
  _Float16* y16[3];
  for (int i = 0; i < 3; ++i) y16[i] = (_Float16*)take(XEL * 2);
  _Float16* attO  = (_Float16*)take(XEL * 2);
  if (off > ws_size) return;  // workspace too small; nothing safe to do

  // 1) casts (U transposed so every GEMM B-fragment is K-contiguous)
  run_cast(q, x16[0], (int)XEL, stream);
  run_cast(k, x16[1], (int)XEL, stream);
  run_cast(v, x16[2], (int)XEL, stream);
  for (int i = 0; i < 4; ++i) {
    transpose_cast_f32_f16<<<((int)FEL + 255) / 256, 256, 0, stream>>>(
        U[i], UT16[i], D_MODEL, RANK);
    run_cast(V[i], V16[i], (int)FEL, stream);
  }

  // 2) input projections: y = (x @ U) @ V^T + b, stored f16 in [B,T,D]
  for (int i = 0; i < 3; ++i) {
    dim3 g1(RANK / 16, ROWS / 16);
    gemm_wmma16<<<g1, 32, 0, stream>>>(x16[i], D_MODEL,
                                       UT16[i], D_MODEL, D_MODEL,
                                       nullptr, tmp16, nullptr, RANK);
    dim3 g2(D_MODEL / 16, ROWS / 16);
    gemm_wmma16<<<g2, 32, 0, stream>>>(tmp16, RANK,
                                       V16[i], RANK, RANK,
                                       bia[i], y16[i], nullptr, D_MODEL);
  }

  // 3) fused attention (scores WMMA -> histogram top-64 -> softmax -> sparse AV)
  {
    const int blocks = BATCH * N_HEADS * (SEQLEN / 16);  // 4096
    const size_t lds = (size_t)16 * SEQLEN * 2;          // 65536 B
    attn_topk_kernel<<<blocks, 256, lds, stream>>>(y16[0], y16[1], y16[2], attO);
  }

  // 4) output projection -> f32 d_out
  {
    dim3 g1(RANK / 16, ROWS / 16);
    gemm_wmma16<<<g1, 32, 0, stream>>>(attO, D_MODEL,
                                       UT16[3], D_MODEL, D_MODEL,
                                       nullptr, tmp16, nullptr, RANK);
    dim3 g2(D_MODEL / 16, ROWS / 16);
    gemm_wmma16<<<g2, 32, 0, stream>>>(tmp16, RANK,
                                       V16[3], RANK, RANK,
                                       bia[3], nullptr, out, D_MODEL);
  }
}
